// DynamicHighCapsule_68195490726086
// MI455X (gfx1250) — matrix operand order repaired and checked by reading, not compiled, not run on hardware
//
#include <hip/hip_runtime.h>
#include <math.h>

// Problem constants (match reference)
#define B_TOT 32
#define IC    2048   // IN_CAPS
#define IE    32     // IN_DIM  (K of the per-i GEMM)
#define OC    64     // OUT_CAPS
#define OD    32     // OUT_DIM
#define BG    8      // b's handled per workgroup
#define NBG   (B_TOT / BG)   // 4 b-groups
#define WAVES 8
#define THREADS (WAVES * 32)

#define TILE_W    (IE * OD)                  // 1024 f32 = 4KB weight tile per (o,i)
#define SMEM_CORE (BG * IC + 2 * BG * OD + BG)
#define SMEM_WORDS (SMEM_CORE + WAVES * 2 * TILE_W)

typedef __attribute__((ext_vector_type(16))) __bf16 v16bf;
typedef __attribute__((ext_vector_type(8)))  float  v8f;
typedef __attribute__((ext_vector_type(4)))  unsigned int u32x4;
typedef __attribute__((ext_vector_type(8)))  int          i32x8;
typedef __attribute__((ext_vector_type(4)))  int          i32x4;

__device__ __forceinline__ v8f wmma_bf16(v16bf a, v16bf b, v8f c) {
  // D = A(16x32 bf16) * B(32x16 bf16) + C(16x16 f32)
  return __builtin_amdgcn_wmma_f32_16x16x32_bf16(false, a, false, b, (short)0, c,
                                                 false, false);
}

// TDM: fetch one 4KB weight tile (1024 f32, contiguous) into LDS.
// o,i are wave-uniform -> descriptor lives in SGPRs; tracked by TENSORcnt.
__device__ __forceinline__ void tdm_fetch_w_tile(const float* gsrc, const float* lds_dst) {
  // Retire any in-flight LDS reads of the staging buffers before TDM overwrites.
  asm volatile("s_wait_dscnt 0x0" ::: "memory");
  unsigned long long ga = (unsigned long long)(uintptr_t)gsrc;
  unsigned int lofs = (unsigned int)(uintptr_t)lds_dst;   // flat LDS addr[31:0] == LDS byte offset
  u32x4 g0;
  g0[0] = 1u;                                              // count=1 valid descriptor
  g0[1] = lofs;                                            // lds_addr (bytes)
  g0[2] = (unsigned int)ga;                                // global_addr[31:0]
  g0[3] = ((unsigned int)(ga >> 32) & 0x01FFFFFFu) | (2u << 30); // addr[56:32] | type=2
  i32x8 g1;
  g1[0] = (int)(2u << 16);        // workgroup_mask=0, data_size=2 (4 bytes), no flags
  g1[1] = (int)(1024u << 16);     // atomic_barrier_addr=0 | tensor_dim0[15:0]=1024
  g1[2] = (int)(1u << 16);        // tensor_dim0[31:16]=0  | tensor_dim1[15:0]=1
  g1[3] = (int)(1024u << 16);     // tensor_dim1[31:16]=0  | tile_dim0=1024
  g1[4] = 0;                      // tile_dim1=0 (unused), tile_dim2=0
  g1[5] = 1024;                   // tensor_dim0_stride[31:0]=1024
  g1[6] = 0;                      // stride hi | tensor_dim1_stride lo
  g1[7] = 0;
  i32x4 gz4  = {0, 0, 0, 0};      // groups 2/3 unused (tile_dim3/4 = 0)
  i32x8 gz8  = {0, 0, 0, 0, 0, 0, 0, 0};
  __builtin_amdgcn_tensor_load_to_lds(g0, g1, gz4, gz4, gz8, 0);
}

__device__ __forceinline__ void tdm_wait_prev() {
  // Double buffered: newest fetch may stay outstanding; oldest must be done.
  asm volatile("s_wait_tensorcnt 0x1" ::: "memory");
}

// A-matrix 16x32 bf16 layout (wave32):
//   lanes 0-15 : row M=lane,    elem 0..15 -> K=kb..kb+15
//   kb = (lane>>4)*16
__device__ __forceinline__ v16bf load_a_tile(const float* __restrict__ x,
                                             int b0, int arow, int i, int kb) {
  v16bf a;
  if (arow < BG) {
    const float* p = x + ((size_t)(b0 + arow) * IC + i) * IE;
    float4 f0 = *(const float4*)(p + kb);
    float4 f1 = *(const float4*)(p + kb + 4);
    float4 f2 = *(const float4*)(p + kb + 8);
    float4 f3 = *(const float4*)(p + kb + 12);
    a[0]=(__bf16)f0.x;  a[1]=(__bf16)f0.y;  a[2]=(__bf16)f0.z;  a[3]=(__bf16)f0.w;
    a[4]=(__bf16)f1.x;  a[5]=(__bf16)f1.y;  a[6]=(__bf16)f1.z;  a[7]=(__bf16)f1.w;
    a[8]=(__bf16)f2.x;  a[9]=(__bf16)f2.y;  a[10]=(__bf16)f2.z; a[11]=(__bf16)f2.w;
    a[12]=(__bf16)f3.x; a[13]=(__bf16)f3.y; a[14]=(__bf16)f3.z; a[15]=(__bf16)f3.w;
  } else {
#pragma unroll
    for (int e = 0; e < 16; ++e) a[e] = (__bf16)0.0f;
  }
  return a;
}

// B-matrix 32x16 bf16 from the LDS-staged f32 tile (row-major [e][d]).
// lanes 0-15: col=lane, K=kb..kb+15 ; lanes 16-31: col=lane-16, K=16+...
// Column reads: stride 32 f32 -> 32 distinct LDS banks, conflict-free.
__device__ __forceinline__ v16bf b_from_stage(const float* sw, int kb, int col) {
  v16bf b;
#pragma unroll
  for (int e = 0; e < 16; ++e) b[e] = (__bf16)sw[(kb + e) * OD + col];
  return b;
}

__device__ __forceinline__ float rsum16(float v) {
  v += __shfl_xor(v, 1, 16);
  v += __shfl_xor(v, 2, 16);
  v += __shfl_xor(v, 4, 16);
  v += __shfl_xor(v, 8, 16);
  return v;
}
__device__ __forceinline__ float rsum32(float v) {
  v += __shfl_xor(v, 1, 32);
  v += __shfl_xor(v, 2, 32);
  v += __shfl_xor(v, 4, 32);
  v += __shfl_xor(v, 8, 32);
  v += __shfl_xor(v, 16, 32);
  return v;
}

// One workgroup = (o, group of BG b's). Priors are recomputed via WMMA on each
// of the 3 routing passes (TDM streams W tiles into LDS; routing state in LDS).
__global__ __launch_bounds__(THREADS)
void caps_routing_kernel(const float* __restrict__ x,
                         const float* __restrict__ Wp,
                         float* __restrict__ out) {
  extern __shared__ float smem[];
  float* s_c     = smem;                    // [BG][IC]  routing logits c1
  float* s_acc   = s_c + BG * IC;           // [BG][OD]  sum / E accumulator
  float* s_out   = s_acc + BG * OD;         // [BG][OD]  squashed out vector
  float* s_z     = s_out + BG * OD;         // [BG]      softmax partition Z
  float* s_stage = smem + SMEM_CORE;        // [WAVES][2][TILE_W] TDM staging

  const int tid  = threadIdx.x;
  const int wave = tid >> 5;
  const int lane = tid & 31;
  const int row  = lane & 15;
  const int h    = lane >> 4;
  const int o    = (int)blockIdx.x >> 2;    // 4 consecutive blocks share o -> L2 reuse
  const int b0   = ((int)blockIdx.x & 3) * BG;

  const int kbA  = h * 16;
  const int kbB  = h * 16;
  const int dcol = row;

  float* my_stage = s_stage + wave * 2 * TILE_W;
  const float* wbase = Wp + (size_t)o * IC * TILE_W;

  for (int t = tid; t < BG * OD; t += THREADS) s_acc[t] = 0.0f;
  if (tid < BG) s_z[tid] = 0.0f;
  __syncthreads();

  // ---------------- Pass 1: c = 0 -> uniform probs; sum_i P via WMMA chaining
  {
    v8f acc0 = {}; v8f acc1 = {};
    tdm_fetch_w_tile(wbase + (size_t)wave * TILE_W, my_stage);          // prologue
    int buf = 0;
    for (int i = wave; i < IC; i += WAVES) {
      int inext = (i + WAVES < IC) ? (i + WAVES) : i;                    // always issue
      tdm_fetch_w_tile(wbase + (size_t)inext * TILE_W, my_stage + (buf ^ 1) * TILE_W);
      tdm_wait_prev();
      const float* sw = my_stage + buf * TILE_W;
      v16bf a   = load_a_tile(x, b0, row, i, kbA);
      v16bf bt0 = b_from_stage(sw, kbB, dcol);
      v16bf bt1 = b_from_stage(sw, kbB, dcol + 16);
      acc0 = wmma_bf16(a, bt0, acc0);
      acc1 = wmma_bf16(a, bt1, acc1);
      buf ^= 1;
    }
    if (h == 0) {                       // only rows m=r (0..7) live (BG=8)
#pragma unroll
      for (int r = 0; r < 8; ++r) {
        atomicAdd(&s_acc[r * OD + dcol],      acc0[r]);
        atomicAdd(&s_acc[r * OD + dcol + 16], acc1[r]);
      }
    }
  }
  __syncthreads();
  if (wave == 0) {                      // out_a = squash(mean_i P, over d)
    const float invN = 1.0f / (float)IC;
    for (int m = 0; m < BG; ++m) {
      float v  = s_acc[m * OD + lane] * invN;
      float sq = rsum32(v * v);
      float n  = sqrtf(sq);
      s_out[m * OD + lane] = v * (n / (1.0f + sq));
      s_acc[m * OD + lane] = 0.0f;
    }
    if (lane < BG) s_z[lane] = 0.0f;
  }
  __syncthreads();

  // ---------------- Pass 2: c1 = P.out_a ; E = sum e^c1 P ; Z = sum e^c1
  {
    float o0[8], o1[8];
#pragma unroll
    for (int r = 0; r < 8; ++r) {
      o0[r] = s_out[r * OD + dcol];
      o1[r] = s_out[r * OD + dcol + 16];
    }
    v8f e0 = {}; v8f e1 = {};
    float zac[8] = {0, 0, 0, 0, 0, 0, 0, 0};
    tdm_fetch_w_tile(wbase + (size_t)wave * TILE_W, my_stage);
    int buf = 0;
    for (int i = wave; i < IC; i += WAVES) {
      int inext = (i + WAVES < IC) ? (i + WAVES) : i;
      tdm_fetch_w_tile(wbase + (size_t)inext * TILE_W, my_stage + (buf ^ 1) * TILE_W);
      tdm_wait_prev();
      const float* sw = my_stage + buf * TILE_W;
      v16bf a   = load_a_tile(x, b0, row, i, kbA);
      v16bf bt0 = b_from_stage(sw, kbB, dcol);
      v16bf bt1 = b_from_stage(sw, kbB, dcol + 16);
      v8f zero = {};
      v8f p0 = wmma_bf16(a, bt0, zero);
      v8f p1 = wmma_bf16(a, bt1, zero);
#pragma unroll
      for (int r = 0; r < 8; ++r) {
        float c1 = rsum16(p0[r] * o0[r] + p1[r] * o1[r]);   // dot over d (32)
        if (lane == 0) s_c[r * IC + i] = c1;
        float w = __expf(c1);
        e0[r]  += w * p0[r];
        e1[r]  += w * p1[r];
        zac[r] += w;
      }
      buf ^= 1;
    }
    if (h == 0) {
#pragma unroll
      for (int r = 0; r < 8; ++r) {
        atomicAdd(&s_acc[r * OD + dcol],      e0[r]);
        atomicAdd(&s_acc[r * OD + dcol + 16], e1[r]);
      }
      if (lane == 0)
#pragma unroll
        for (int r = 0; r < 8; ++r) atomicAdd(&s_z[r], zac[r]);
    }
  }
  __syncthreads();
  if (wave == 0) {                      // out_b = squash(E/Z)
    for (int m = 0; m < BG; ++m) {
      float v  = s_acc[m * OD + lane] / s_z[m];
      float sq = rsum32(v * v);
      float n  = sqrtf(sq);
      s_out[m * OD + lane] = v * (n / (1.0f + sq));
      s_acc[m * OD + lane] = 0.0f;
    }
    if (lane < BG) s_z[lane] = 0.0f;
  }
  __syncthreads();

  // ---------------- Pass 3: c2 = c1 + P.out_b ; final E/Z -> squash -> output
  {
    float o0[8], o1[8];
#pragma unroll
    for (int r = 0; r < 8; ++r) {
      o0[r] = s_out[r * OD + dcol];
      o1[r] = s_out[r * OD + dcol + 16];
    }
    v8f e0 = {}; v8f e1 = {};
    float zac[8] = {0, 0, 0, 0, 0, 0, 0, 0};
    tdm_fetch_w_tile(wbase + (size_t)wave * TILE_W, my_stage);
    int buf = 0;
    for (int i = wave; i < IC; i += WAVES) {
      int inext = (i + WAVES < IC) ? (i + WAVES) : i;
      tdm_fetch_w_tile(wbase + (size_t)inext * TILE_W, my_stage + (buf ^ 1) * TILE_W);
      tdm_wait_prev();
      const float* sw = my_stage + buf * TILE_W;
      v16bf a   = load_a_tile(x, b0, row, i, kbA);
      v16bf bt0 = b_from_stage(sw, kbB, dcol);
      v16bf bt1 = b_from_stage(sw, kbB, dcol + 16);
      v8f zero = {};
      v8f p0 = wmma_bf16(a, bt0, zero);
      v8f p1 = wmma_bf16(a, bt1, zero);
#pragma unroll
      for (int r = 0; r < 8; ++r) {
        float c2 = s_c[r * IC + i] + rsum16(p0[r] * o0[r] + p1[r] * o1[r]);
        float w  = __expf(c2);
        e0[r]  += w * p0[r];
        e1[r]  += w * p1[r];
        zac[r] += w;
      }
      buf ^= 1;
    }
    if (h == 0) {
#pragma unroll
      for (int r = 0; r < 8; ++r) {
        atomicAdd(&s_acc[r * OD + dcol],      e0[r]);
        atomicAdd(&s_acc[r * OD + dcol + 16], e1[r]);
      }
      if (lane == 0)
#pragma unroll
        for (int r = 0; r < 8; ++r) atomicAdd(&s_z[r], zac[r]);
    }
  }
  __syncthreads();
  if (wave == 0) {                      // result = squash(E/Z) -> d_out[b,o,d]
    for (int m = 0; m < BG; ++m) {
      float v  = s_acc[m * OD + lane] / s_z[m];
      float sq = rsum32(v * v);
      float n  = sqrtf(sq);
      out[((size_t)(b0 + m) * OC + o) * OD + lane] = v * (n / (1.0f + sq));
    }
  }
}

extern "C" void kernel_launch(void* const* d_in, const int* in_sizes, int n_in,
                              void* d_out, int out_size, void* d_ws, size_t ws_size,
                              hipStream_t stream) {
  (void)in_sizes; (void)n_in; (void)d_ws; (void)ws_size; (void)out_size;
  const float* x  = (const float*)d_in[0];   // [32, 2048, 32] f32
  const float* Wp = (const float*)d_in[1];   // [64, 2048, 32, 32] f32
  float* out = (float*)d_out;                // [32, 64, 32] f32

  const size_t shmem = (size_t)SMEM_WORDS * sizeof(float); // ~130 KB (fits 320 KB WGP LDS)
  caps_routing_kernel<<<dim3(OC * NBG), dim3(THREADS), shmem, stream>>>(x, Wp, out);
}
